// PairwiseAttention_84344567759563
// MI455X (gfx1250) — matrix operand order repaired and checked by reading, not compiled, not run on hardware
//
#include <hip/hip_runtime.h>
#include <hip/hip_bf16.h>
#include <math.h>

typedef _Float16 f16;
typedef _Float16 v16h __attribute__((ext_vector_type(16)));
typedef _Float16 h8   __attribute__((ext_vector_type(8)));
typedef float    v8f  __attribute__((ext_vector_type(8)));

#define DEV __device__ __forceinline__

DEV v8f wmma32(v16h a, v16h b, v8f c) {
  return __builtin_amdgcn_wmma_f32_16x16x32_f16(false, a, false, b, (short)0, c, false, false);
}

// A-operand fragment (16x32 f16): lane half g holds K runs [8g,8g+8) and [16+8g,16+8g+8)
DEV v16h frag_a(const f16* rowk, int g) {
  h8 lo = *(const h8*)(rowk + 8 * g);
  h8 hi = *(const h8*)(rowk + 16 + 8 * g);
  v16h r;
#pragma unroll
  for (int i = 0; i < 8; i++) { r[i] = lo[i]; r[8 + i] = hi[i]; }
  return r;
}
// B-operand fragment (32x16 f16): lane half g holds K runs [8g,8g+8) and [16+8g,16+8g+8)
DEV v16h frag_b(const f16* rowk, int g) {
  h8 lo = *(const h8*)(rowk + 8 * g);
  h8 hi = *(const h8*)(rowk + 16 + 8 * g);
  v16h r;
#pragma unroll
  for (int i = 0; i < 8; i++) { r[i] = lo[i]; r[8 + i] = hi[i]; }
  return r;
}

DEV float sigm(float x) { return 1.f / (1.f + __expf(-x)); }

template <int NT>
DEV void gemm_acc(const f16* A, const f16* BT, int K, int m0, int n0, int lane, v8f* acc) {
  int g = lane >> 4, l = lane & 15;
  const f16* ar = A + (size_t)(m0 + l) * K;
  const f16* br = BT + (size_t)(n0 + l) * K;
  for (int k0 = 0; k0 < K; k0 += 32) {
    v16h a = frag_a(ar + k0, g);
#pragma unroll
    for (int t = 0; t < NT; t++) {
      v16h b = frag_b(br + (size_t)(t * 16) * K + k0, g);
      acc[t] = wmma32(a, b, acc[t]);
    }
  }
}

// ---------------- weight transpose fp32[K,N] -> f16[N,K] ----------------
__global__ void k_wt(const float* __restrict__ W, f16* __restrict__ WT, int K, int N) {
  int idx = blockIdx.x * 256 + threadIdx.x;
  if (idx >= K * N) return;
  int n = idx / K, k = idx % K;
  WT[idx] = (f16)W[(size_t)k * N + n];
}

// ---------------- LayerNorm over last dim (128) -> f16 ----------------
__global__ void k_ln(const float* __restrict__ X, const float* __restrict__ gw,
                     const float* __restrict__ bw, f16* __restrict__ out) {
  __shared__ float red[128];
  int row = blockIdx.x, t = threadIdx.x;
  float v = X[(size_t)row * 128 + t];
  red[t] = v; __syncthreads();
  for (int s = 64; s; s >>= 1) { if (t < s) red[t] += red[t + s]; __syncthreads(); }
  float mean = red[0] * (1.f / 128.f);
  __syncthreads();
  float d = v - mean;
  red[t] = d * d; __syncthreads();
  for (int s = 64; s; s >>= 1) { if (t < s) red[t] += red[t + s]; __syncthreads(); }
  float var = red[0] * (1.f / 128.f);
  float r = rsqrtf(var + 1e-5f);
  out[(size_t)row * 128 + t] = (f16)(d * r * gw[t] + bw[t]);
}

// ---------------- edge bias: bias[h,m] = sum_d x[m,d]*We[d,h] ----------------
__global__ void k_bias(const float* __restrict__ X, const float* __restrict__ We,
                       float* __restrict__ B) {
  int t = blockIdx.x * 256 + threadIdx.x;
  int m = t >> 3, h = t & 7;
  float acc = 0.f;
  for (int d = 0; d < 128; d++) acc += X[(size_t)m * 128 + d] * We[d * 8 + h];
  B[(size_t)h * 65536 + m] = acc;
}

// ---------------- gated projection: out_d_plane = (A@Wp+bp)*sigmoid(A@Wg+bg), d-major ----------------
__global__ __launch_bounds__(256) void k_gemm_gated(
    const f16* __restrict__ A, const f16* __restrict__ BTp, const float* __restrict__ bp,
    const f16* __restrict__ BTg, const float* __restrict__ bg,
    f16* __restrict__ outp, int trans) {
  int lane = threadIdx.x & 31, wave = threadIdx.x >> 5;
  int g = lane >> 4, l = lane & 15;
  int m0 = blockIdx.y * 128 + wave * 16;
  int n0 = blockIdx.x * 64;
  v8f accp[4] = {}, accg[4] = {};
  const f16* ar = A + (size_t)(m0 + l) * 128;
  const f16* brp = BTp + (size_t)(n0 + l) * 128;
  const f16* brg = BTg + (size_t)(n0 + l) * 128;
  for (int k0 = 0; k0 < 128; k0 += 32) {
    v16h a = frag_a(ar + k0, g);
#pragma unroll
    for (int t = 0; t < 4; t++) {
      v16h b1 = frag_b(brp + (size_t)(t * 16) * 128 + k0, g);
      accp[t] = wmma32(a, b1, accp[t]);
      v16h b2 = frag_b(brg + (size_t)(t * 16) * 128 + k0, g);
      accg[t] = wmma32(a, b2, accg[t]);
    }
  }
#pragma unroll
  for (int t = 0; t < 4; t++)
#pragma unroll
    for (int e = 0; e < 8; e++) {
      int m = m0 + e + 8 * g;
      int d = n0 + t * 16 + l;
      float v = (accp[t][e] + bp[d]) * sigm(accg[t][e] + bg[d]);
      size_t plane = trans ? (size_t)(m & 255) * 256 + (size_t)(m >> 8) : (size_t)m;
      outp[(size_t)d * 65536 + plane] = (f16)v;
    }
}

// ---------------- sigmoid(A@W + b) -> f32 [m,128] ----------------
__global__ __launch_bounds__(256) void k_gemm_sig_f32(
    const f16* __restrict__ A, const f16* __restrict__ BT,
    const float* __restrict__ bias, float* __restrict__ out) {
  int lane = threadIdx.x & 31, wave = threadIdx.x >> 5;
  int g = lane >> 4, l = lane & 15;
  int m0 = blockIdx.y * 128 + wave * 16;
  int n0 = blockIdx.x * 64;
  v8f acc[4] = {};
  gemm_acc<4>(A, BT, 128, m0, n0, lane, acc);
#pragma unroll
  for (int t = 0; t < 4; t++)
#pragma unroll
    for (int e = 0; e < 8; e++) {
      int m = m0 + e + 8 * g;
      int d = n0 + t * 16 + l;
      out[(size_t)m * 128 + d] = sigm(acc[t][e] + bias[d]);
    }
}

// ---------------- plain A@W * scale -> f16 [m,Nout] ----------------
__global__ __launch_bounds__(256) void k_gemm_plain(
    const f16* __restrict__ A, const f16* __restrict__ BT,
    f16* __restrict__ out, int K, int Nout, float scale) {
  int lane = threadIdx.x & 31, wave = threadIdx.x >> 5;
  int g = lane >> 4, l = lane & 15;
  int m0 = blockIdx.y * 128 + wave * 16;
  int n0 = blockIdx.x * 64;
  v8f acc[4] = {};
  gemm_acc<4>(A, BT, K, m0, n0, lane, acc);
#pragma unroll
  for (int t = 0; t < 4; t++)
#pragma unroll
    for (int e = 0; e < 8; e++) {
      int m = m0 + e + 8 * g;
      int n = n0 + t * 16 + l;
      out[(size_t)m * Nout + n] = (f16)(acc[t][e] * scale);
    }
}

// ---------------- sigmoid(A@W + b) -> f16 [m,256] ----------------
__global__ __launch_bounds__(256) void k_gemm_sig16(
    const f16* __restrict__ A, const f16* __restrict__ BT,
    const float* __restrict__ bias, f16* __restrict__ out) {
  int lane = threadIdx.x & 31, wave = threadIdx.x >> 5;
  int g = lane >> 4, l = lane & 15;
  int m0 = blockIdx.y * 128 + wave * 16;
  int n0 = blockIdx.x * 64;
  v8f acc[4] = {};
  gemm_acc<4>(A, BT, 128, m0, n0, lane, acc);
#pragma unroll
  for (int t = 0; t < 4; t++)
#pragma unroll
    for (int e = 0; e < 8; e++) {
      int m = m0 + e + 8 * g;
      int n = n0 + t * 16 + l;
      out[(size_t)m * 256 + n] = (f16)sigm(acc[t][e] + bias[n]);
    }
}

// ---------------- triangle einsum per channel d: OUT[i,j] = sum_k A[i,k]B[j,k] ----------------
__global__ __launch_bounds__(256) void k_tri_einsum(const f16* __restrict__ Ap,
                                                    const f16* __restrict__ Bp,
                                                    float* __restrict__ TRI) {
  int d = blockIdx.z;
  const f16* A = Ap + (size_t)d * 65536;
  const f16* B = Bp + (size_t)d * 65536;
  int lane = threadIdx.x & 31, wave = threadIdx.x >> 5;
  int g = lane >> 4, l = lane & 15;
  int m0 = blockIdx.y * 128 + wave * 16;
  int n0 = blockIdx.x * 64;
  v8f acc[4] = {};
  const f16* ar = A + (size_t)(m0 + l) * 256;
  const f16* br = B + (size_t)(n0 + l) * 256;
  for (int k0 = 0; k0 < 256; k0 += 32) {
    v16h a = frag_a(ar + k0, g);
#pragma unroll
    for (int t = 0; t < 4; t++) {
      v16h bb = frag_b(br + (size_t)(t * 16) * 256 + k0, g);
      acc[t] = wmma32(a, bb, acc[t]);
    }
  }
#pragma unroll
  for (int t = 0; t < 4; t++)
#pragma unroll
    for (int e = 0; e < 8; e++) {
      int i = m0 + e + 8 * g;
      int j = n0 + t * 16 + l;
      TRI[((size_t)i * 256 + j) * 128 + d] = acc[t][e];
    }
}

// ---------------- triangle final: XC = (A@Wo+bo)*OG + XC ----------------
__global__ __launch_bounds__(256) void k_gemm_trifinal(
    const f16* __restrict__ A, const f16* __restrict__ BT, const float* __restrict__ bias,
    const float* __restrict__ OG, float* __restrict__ XC) {
  int lane = threadIdx.x & 31, wave = threadIdx.x >> 5;
  int g = lane >> 4, l = lane & 15;
  int m0 = blockIdx.y * 128 + wave * 16;
  int n0 = blockIdx.x * 64;
  v8f acc[4] = {};
  gemm_acc<4>(A, BT, 128, m0, n0, lane, acc);
#pragma unroll
  for (int t = 0; t < 4; t++)
#pragma unroll
    for (int e = 0; e < 8; e++) {
      int m = m0 + e + 8 * g;
      int d = n0 + t * 16 + l;
      size_t idx = (size_t)m * 128 + d;
      XC[idx] = (acc[t][e] + bias[d]) * OG[idx] + XC[idx];
    }
}

// ---------------- attention out-proj: OUT = A@Wo + bo + XC ----------------
__global__ __launch_bounds__(256) void k_gemm_outproj(
    const f16* __restrict__ A, const f16* __restrict__ BT, const float* __restrict__ bias,
    const float* __restrict__ XC, float* __restrict__ OUT) {
  int lane = threadIdx.x & 31, wave = threadIdx.x >> 5;
  int g = lane >> 4, l = lane & 15;
  int m0 = blockIdx.y * 128 + wave * 16;
  int n0 = blockIdx.x * 64;
  v8f acc[4] = {};
  gemm_acc<4>(A, BT, 256, m0, n0, lane, acc);
#pragma unroll
  for (int t = 0; t < 4; t++)
#pragma unroll
    for (int e = 0; e < 8; e++) {
      int m = m0 + e + 8 * g;
      int d = n0 + t * 16 + l;
      size_t idx = (size_t)m * 128 + d;
      OUT[idx] = acc[t][e] + bias[d] + XC[idx];
    }
}

// ---------------- fused axial attention core per (b,h) ----------------
__global__ __launch_bounds__(128) void k_attn(
    const f16* __restrict__ Q, const f16* __restrict__ Km, const f16* __restrict__ Vm,
    const f16* __restrict__ G, const float* __restrict__ BIAS,
    f16* __restrict__ AOUT, int row_mode) {
  __shared__ __align__(16) f16 Ksh[256 * 32];
  __shared__ __align__(16) f16 VTsh[32 * 256];
  __shared__ __align__(16) f16 Psh[4 * 16 * 256];
  int b = blockIdx.z, h = blockIdx.y;
  for (int idx = threadIdx.x; idx < 256 * 32; idx += 128) {
    int j = idx >> 5, dh = idx & 31;
    size_t m = row_mode ? (size_t)b * 256 + j : (size_t)j * 256 + b;
    f16 kv = Km[m * 256 + h * 32 + dh];
    f16 vv = Vm[m * 256 + h * 32 + dh];
    Ksh[j * 32 + dh] = kv;
    VTsh[dh * 256 + j] = vv;
  }
  __syncthreads();
  int lane = threadIdx.x & 31, wave = threadIdx.x >> 5;
  int g = lane >> 4, l = lane & 15;
  int q0 = blockIdx.x * 64 + wave * 16;
  size_t mq = row_mode ? (size_t)b * 256 + (q0 + l) : (size_t)(q0 + l) * 256 + b;
  v16h aq = frag_a(Q + mq * 256 + h * 32, g);
  v8f s[16];
#pragma unroll
  for (int jc = 0; jc < 16; jc++) {
    v16h bk = frag_b(Ksh + (jc * 16 + l) * 32, g);
    v8f z = {};
    s[jc] = wmma32(aq, bk, z);
  }
  float rmax[8], rsum[8];
#pragma unroll
  for (int e = 0; e < 8; e++) rmax[e] = -1e30f;
  const float* bb = BIAS + (size_t)h * 65536;
#pragma unroll
  for (int jc = 0; jc < 16; jc++)
#pragma unroll
    for (int e = 0; e < 8; e++) {
      int qi = q0 + e + 8 * g;
      float v = s[jc][e] + bb[(size_t)qi * 256 + jc * 16 + l];
      s[jc][e] = v;
      rmax[e] = fmaxf(rmax[e], v);
    }
#pragma unroll
  for (int e = 0; e < 8; e++) {
    float m = rmax[e];
    m = fmaxf(m, __shfl_xor(m, 1, 32));
    m = fmaxf(m, __shfl_xor(m, 2, 32));
    m = fmaxf(m, __shfl_xor(m, 4, 32));
    m = fmaxf(m, __shfl_xor(m, 8, 32));
    rmax[e] = m;
    rsum[e] = 0.f;
  }
#pragma unroll
  for (int jc = 0; jc < 16; jc++)
#pragma unroll
    for (int e = 0; e < 8; e++) {
      float p = __expf(s[jc][e] - rmax[e]);
      s[jc][e] = p;
      rsum[e] += p;
    }
#pragma unroll
  for (int e = 0; e < 8; e++) {
    float t = rsum[e];
    t += __shfl_xor(t, 1, 32);
    t += __shfl_xor(t, 2, 32);
    t += __shfl_xor(t, 4, 32);
    t += __shfl_xor(t, 8, 32);
    rsum[e] = 1.f / t;
  }
  f16* pw = Psh + wave * 16 * 256;
#pragma unroll
  for (int jc = 0; jc < 16; jc++)
#pragma unroll
    for (int e = 0; e < 8; e++)
      pw[(e + 8 * g) * 256 + jc * 16 + l] = (f16)s[jc][e];
  __syncthreads();
#pragma unroll
  for (int nc = 0; nc < 2; nc++) {
    v8f o = {};
#pragma unroll
    for (int kc = 0; kc < 8; kc++) {
      v16h ap = frag_a(pw + l * 256 + kc * 32, g);
      v16h bv = frag_b(VTsh + (nc * 16 + l) * 256 + kc * 32, g);
      o = wmma32(ap, bv, o);
    }
#pragma unroll
    for (int e = 0; e < 8; e++) {
      int qi = q0 + e + 8 * g;
      int dh = nc * 16 + l;
      size_t m = row_mode ? (size_t)b * 256 + qi : (size_t)qi * 256 + b;
      float val = o[e] * rsum[e];
      val *= (float)G[m * 256 + h * 32 + dh];
      AOUT[m * 256 + h * 32 + dh] = (f16)val;
    }
  }
}

// =================== host orchestration ===================
struct TriP { const float *ng,*nb,*ong,*onb,*Wl,*bl,*Wr,*br,*Wlg,*blg,*Wrg,*brg,*Wog,*bog,*Wo,*bo; };
struct AxP  { const float *ng,*nb,*We,*Wq,*Wkv,*Wg,*bg,*Wo,*bo; };

static void run_tri(const TriP& P, f16* const* Wt, float* XCUR, f16* XN,
                    f16* LT, f16* RT, float* OG, float* TRI, int outgoing, hipStream_t s) {
  k_ln<<<65536, 128, 0, s>>>(XCUR, P.ng, P.nb, XN);
  int tr = outgoing ? 0 : 1;
  k_gemm_gated<<<dim3(2, 512), 256, 0, s>>>(XN, Wt[0], P.bl, Wt[2], P.blg, LT, tr);
  k_gemm_gated<<<dim3(2, 512), 256, 0, s>>>(XN, Wt[1], P.br, Wt[3], P.brg, RT, tr);
  k_gemm_sig_f32<<<dim3(2, 512), 256, 0, s>>>(XN, Wt[4], P.bog, OG);
  const f16* Aein = outgoing ? LT : RT;
  const f16* Bein = outgoing ? RT : LT;
  k_tri_einsum<<<dim3(4, 2, 128), 256, 0, s>>>(Aein, Bein, TRI);
  k_ln<<<65536, 128, 0, s>>>(TRI, P.ong, P.onb, XN);
  k_gemm_trifinal<<<dim3(2, 512), 256, 0, s>>>(XN, Wt[5], P.bo, OG, XCUR);
}

static void run_ax(const AxP& P, f16* const* Wt, float* XCUR, f16* XN,
                   f16* Qb, f16* Kb, f16* Vb, f16* Gb, float* BIAS, f16* AOUT,
                   int row_mode, float* DST, hipStream_t s) {
  k_bias<<<2048, 256, 0, s>>>(XCUR, P.We, BIAS);
  k_ln<<<65536, 128, 0, s>>>(XCUR, P.ng, P.nb, XN);
  k_gemm_plain<<<dim3(4, 512), 256, 0, s>>>(XN, Wt[0], Qb, 128, 256, 0.17677669529663687f);
  k_gemm_plain<<<dim3(4, 512), 256, 0, s>>>(XN, Wt[1], Kb, 128, 256, 1.f);
  k_gemm_plain<<<dim3(4, 512), 256, 0, s>>>(XN, Wt[1] + (size_t)256 * 128, Vb, 128, 256, 1.f);
  k_gemm_sig16<<<dim3(4, 512), 256, 0, s>>>(XN, Wt[2], P.bg, Gb);
  k_attn<<<dim3(4, 8, 256), 128, 0, s>>>(Qb, Kb, Vb, Gb, BIAS, AOUT, row_mode);
  k_gemm_outproj<<<dim3(2, 512), 256, 0, s>>>(AOUT, Wt[3], P.bo, XCUR, DST);
}

extern "C" void kernel_launch(void* const* d_in, const int* in_sizes, int n_in,
                              void* d_out, int out_size, void* d_ws, size_t ws_size,
                              hipStream_t stream) {
  (void)in_sizes; (void)n_in; (void)out_size; (void)ws_size;
  auto F = [&](int i) { return (const float*)d_in[i]; };
  TriP t0{F(1),F(2),F(3),F(4),F(5),F(6),F(7),F(8),F(9),F(10),F(11),F(12),F(13),F(14),F(15),F(16)};
  TriP t1{F(17),F(18),F(19),F(20),F(21),F(22),F(23),F(24),F(25),F(26),F(27),F(28),F(29),F(30),F(31),F(32)};
  AxP  a0{F(33),F(34),F(35),F(36),F(37),F(38),F(39),F(40),F(41)};
  AxP  a1{F(42),F(43),F(44),F(45),F(46),F(47),F(48),F(49),F(50)};

  const size_t MB = 1ull << 20;
  char* w = (char*)d_ws;
  float* XCUR = (float*)w;                 // 32 MB fp32 residual stream
  f16*   WTb  = (f16*)(w + 32 * MB);       // 1 MB transposed f16 weights
  f16*   XN   = (f16*)(w + 33 * MB);       // 16 MB LN output f16
  char*  SH   = w + 49 * MB;               // shared phase region
  // triangle phase
  f16*   LT   = (f16*)SH;                  // 16 MB d-major left
  f16*   RT   = (f16*)(SH + 16 * MB);      // 16 MB d-major right
  float* OG   = (float*)(SH + 32 * MB);    // 32 MB out gate
  float* TRI  = (float*)(SH + 64 * MB);    // 32 MB triangle result
  // attention phase (aliases triangle region)
  f16*   Qb   = (f16*)SH;                  // 32 MB
  f16*   Kb   = (f16*)(SH + 32 * MB);      // 32 MB
  f16*   Vb   = (f16*)(SH + 64 * MB);      // 32 MB
  f16*   Gb   = (f16*)(SH + 96 * MB);      // 32 MB
  float* BIAS = (float*)(SH + 128 * MB);   // 2 MB
  f16*   AOUT = (f16*)(SH + 131 * MB);     // 32 MB

  auto wt = [&](const float* Wsrc, f16* dst, int K, int N) {
    int tot = K * N;
    k_wt<<<(tot + 255) / 256, 256, 0, stream>>>(Wsrc, dst, K, N);
  };

  f16* p = WTb;
  f16* T0[6]; const float* S0[6] = {t0.Wl, t0.Wr, t0.Wlg, t0.Wrg, t0.Wog, t0.Wo};
  for (int i = 0; i < 6; i++) { T0[i] = p; wt(S0[i], p, 128, 128); p += 16384; }
  f16* T1[6]; const float* S1[6] = {t1.Wl, t1.Wr, t1.Wlg, t1.Wrg, t1.Wog, t1.Wo};
  for (int i = 0; i < 6; i++) { T1[i] = p; wt(S1[i], p, 128, 128); p += 16384; }
  f16* A0[4];
  A0[0] = p; wt(a0.Wq,  p, 128, 256); p += 32768;
  A0[1] = p; wt(a0.Wkv, p, 128, 512); p += 65536;
  A0[2] = p; wt(a0.Wg,  p, 128, 256); p += 32768;
  A0[3] = p; wt(a0.Wo,  p, 256, 128); p += 32768;
  f16* A1[4];
  A1[0] = p; wt(a1.Wq,  p, 128, 256); p += 32768;
  A1[1] = p; wt(a1.Wkv, p, 128, 512); p += 65536;
  A1[2] = p; wt(a1.Wg,  p, 128, 256); p += 32768;
  A1[3] = p; wt(a1.Wo,  p, 256, 128); p += 32768;

  hipMemcpyAsync(XCUR, d_in[0], (size_t)65536 * 128 * 4, hipMemcpyDeviceToDevice, stream);

  run_tri(t0, T0, XCUR, XN, LT, RT, OG, TRI, 1, stream);
  run_tri(t1, T1, XCUR, XN, LT, RT, OG, TRI, 0, stream);
  run_ax(a0, A0, XCUR, XN, Qb, Kb, Vb, Gb, BIAS, AOUT, 1, XCUR, stream);
  run_ax(a1, A1, XCUR, XN, Qb, Kb, Vb, Gb, BIAS, AOUT, 0, (float*)d_out, stream);
}
